// NMSRelation_25031069401587
// MI455X (gfx1250) — matrix operand order repaired and checked by reading, not compiled, not run on hardware
//
#include <hip/hip_runtime.h>
#include <hip/hip_bf16.h>
#include <math.h>

#define N_ROIS 256
#define NUM_FG 16
#define FEAT   128
#define GROUPS 256      // NUM_FG * 16
#define HEAD   64
#define DIM0   1024
#define DIM2   128
#define KC     2048     // FC1 * FEAT

typedef __attribute__((ext_vector_type(16))) __bf16 v16bf;
typedef __attribute__((ext_vector_type(8)))  float  v8f;

__device__ __forceinline__ v8f wmma_bf16(v16bf a, v16bf b, v8f c) {
  // 8 args: (neg_a, A, neg_b, B, c_mod, C, reuse_a, reuse_b)
  return __builtin_amdgcn_wmma_f32_16x16x32_bf16(false, a, false, b, (short)0, c,
                                                 false, false);
}

// ---- WMMA per-lane tile loaders (ISA 7.12.2 layouts, wave32) ----

// A tile (16 rows x 32 K) from row-major [row][k] memory.
__device__ __forceinline__ v16bf load_A(const __bf16* base, int stride, int lane) {
  int row = lane & 15, hi = lane >> 4;
  const __bf16* p = base + (size_t)row * stride + hi * 8;
  v16bf a;
#pragma unroll
  for (int h = 0; h < 8; ++h) a[h] = p[h];
#pragma unroll
  for (int h = 0; h < 8; ++h) a[8 + h] = p[16 + h];
  return a;
}

// B tile (32 K x 16 N) where SOURCE is row-major [n][k] (transposed access;
// two 16 B chunks per lane at k = 8*hi and 16 + 8*hi).
__device__ __forceinline__ v16bf load_B_t(const __bf16* base, int stride, int lane) {
  int col = lane & 15, hi = lane >> 4;
  const __bf16* p = base + (size_t)col * stride + hi * 8;
  v16bf b;
#pragma unroll
  for (int h = 0; h < 8; ++h) b[h] = p[h];
#pragma unroll
  for (int h = 0; h < 8; ++h) b[8 + h] = p[16 + h];
  return b;
}

// ---- layout/convert kernels ----

// rfb[f][n][d] = bf16(roi_feat[n][f][d])
__global__ void __launch_bounds__(256) pack_rf(const float* __restrict__ roi,
                                               __bf16* __restrict__ rfb) {
  int i = blockIdx.x * 256 + threadIdx.x;              // 524288
  int f = i >> 15, r = i & 32767, n = r >> 7, d = r & 127;
  rfb[i] = (__bf16)roi[((size_t)n * NUM_FG + f) * FEAT + d];
}

// rfT[f][d][n] = bf16(roi_feat[n][f][d])  (transposed copy for PV B tiles)
__global__ void __launch_bounds__(256) pack_rfT(const float* __restrict__ roi,
                                                __bf16* __restrict__ rfT) {
  int i = blockIdx.x * 256 + threadIdx.x;              // 524288
  int f = i >> 15, r = i & 32767, d = r >> 8, n = r & 255;
  rfT[i] = (__bf16)roi[((size_t)n * NUM_FG + f) * FEAT + d];
}

__global__ void __launch_bounds__(256) cvt_bf16(const float* __restrict__ src,
                                                __bf16* __restrict__ dst, int n) {
  int i = blockIdx.x * 256 + threadIdx.x;
  if (i < n) dst[i] = (__bf16)src[i];
}

// ---- phase B: sin/cos position embedding -> FC1 -> relu -> log, fused ----
__global__ void __launch_bounds__(256) bias_log(const float* __restrict__ pos,
                                                const float* __restrict__ W1,
                                                const float* __restrict__ b1,
                                                float* __restrict__ aff) {
  __shared__ float sW1[1024];
  __shared__ float sb1[16];
  for (int i = threadIdx.x; i < 1024; i += 256) sW1[i] = W1[i];
  if (threadIdx.x < 16) sb1[threadIdx.x] = b1[threadIdx.x];
  __syncthreads();

  int idx = blockIdx.x * 256 + threadIdx.x;            // 16*256*256 threads
  int f = idx >> 16, rem = idx & 65535;
  int n = rem >> 8, m = rem & 255;
  const float* p = pos + ((size_t)(f * N_ROIS + n) * N_ROIS + m) * 2;
  float p0 = p[0] * 100.f, p1 = p[1] * 100.f;

  float acc[16];
#pragma unroll
  for (int j = 0; j < 16; ++j) acc[j] = sb1[j];

#pragma unroll 4
  for (int c = 0; c < 16; ++c) {
    float inv = __powf(1000.f, -(float)c * 0.0625f);   // 1 / 1000^(4c/64)
    float d0 = p0 * inv, d1 = p1 * inv;
    float s0 = __sinf(d0), c0 = __cosf(d0);
    float s1 = __sinf(d1), c1 = __cosf(d1);
#pragma unroll
    for (int j = 0; j < 16; ++j) {
      const float* w = sW1 + j * 64 + c;               // emb order: [s0|c0|s1|c1] x16
      acc[j] += s0 * w[0] + c0 * w[16] + s1 * w[32] + c1 * w[48];
    }
  }
  float* ag = aff + (size_t)f * 16 * 65536 + (size_t)n * 256 + m;
#pragma unroll
  for (int j = 0; j < 16; ++j)
    ag[(size_t)j * 65536] = __logf(fmaxf(acc[j], 1e-6f));
}

// ---- phase A: grouped Q/K projection GEMM (M=4096, N=1024, K=128) ----
__global__ void __launch_bounds__(128) gemm_qk(const __bf16* __restrict__ rfb,
                                               const __bf16* __restrict__ W2b,
                                               const __bf16* __restrict__ W3b,
                                               const float* __restrict__ b2,
                                               const float* __restrict__ b3,
                                               __bf16* __restrict__ qbuf,
                                               __bf16* __restrict__ kbuf) {
  int wave = threadIdx.x >> 5, lane = threadIdx.x & 31;
  int id = blockIdx.x * 4 + wave;                      // 16384 tiles
  int ct = id & 63, nt = (id >> 6) & 15, f = id >> 10;
  int n0 = nt * 16, cb = ct * 16;

  const __bf16* A0 = rfb + ((size_t)f * N_ROIS + n0) * FEAT;
  v8f aq = {}, ak = {};
#pragma unroll
  for (int ks = 0; ks < FEAT; ks += 32) {
    v16bf a  = load_A(A0 + ks, FEAT, lane);
    v16bf tq = load_B_t(W2b + (size_t)cb * FEAT + ks, FEAT, lane);
    aq = wmma_bf16(a, tq, aq);
    v16bf tk = load_B_t(W3b + (size_t)cb * FEAT + ks, FEAT, lane);
    ak = wmma_bf16(a, tk, ak);
  }
  int col = lane & 15, hi = lane >> 4;
  int j = cb >> 6, hb = cb & 63;
  int c = cb + col;
  float biasq = b2[c], biask = b3[c];
  __bf16* qo = qbuf + ((size_t)(f * 16 + j) * N_ROIS + n0) * HEAD + hb;
  __bf16* ko = kbuf + ((size_t)(f * 16 + j) * N_ROIS + n0) * HEAD + hb;
#pragma unroll
  for (int r = 0; r < 8; ++r) {
    int row = r + hi * 8;
    qo[(size_t)row * HEAD + col] = (__bf16)(aq[r] + biasq);
    ko[(size_t)row * HEAD + col] = (__bf16)(ak[r] + biask);
  }
}

// ---- phase C: per-group attention: S=QK^T/8 + logbias; softmax; P@V ----
// One block (2 waves) per (group g, 16-row n tile). K_g is async-staged into
// LDS with GLOBAL_LOAD_ASYNC_TO_LDS_B128 (ASYNCcnt); log-bias is read from the
// aff output region and overwritten in place with the softmax result.
__global__ void __launch_bounds__(64) attn(const __bf16* __restrict__ qbuf,
                                           const __bf16* __restrict__ kbuf,
                                           const __bf16* __restrict__ rfT,
                                           float* __restrict__ aff,
                                           __bf16* __restrict__ avb) {
  __shared__ float  Ssh[16 * 256];     // 16 KB: scores / exp scratch
  __shared__ __bf16 Psh[16 * 256];     //  8 KB: bf16 softmax for PV
  __shared__ __bf16 Ksh[N_ROIS * HEAD];// 32 KB: staged K_g

  int g = blockIdx.x >> 4, nt = blockIdx.x & 15;
  int wave = threadIdx.x >> 5, lane = threadIdx.x & 31;
  int n0 = nt * 16;
  int f = g >> 4, jj = g & 15;

  const __bf16* Qg = qbuf + ((size_t)g * N_ROIS + n0) * HEAD;
  const __bf16* Kg = kbuf + (size_t)g * N_ROIS * HEAD;
  float* affg = aff + (size_t)g * N_ROIS * N_ROIS;

  // ---- async stage K_g (256x64 bf16 = 32 KB) into LDS, 16 B per lane/iter ----
#pragma unroll 4
  for (int it = 0; it < 32; ++it) {
    int e = (it * 64 + threadIdx.x) * 8;               // bf16 element offset
    unsigned lds = (unsigned)(size_t)(Ksh + e);        // low 32b of generic = LDS addr
    const __bf16* ga = Kg + e;
    asm volatile("global_load_async_to_lds_b128 %0, %1, off"
                 :: "v"(lds), "v"(ga) : "memory");
  }
  asm volatile("s_wait_asynccnt 0x0" ::: "memory");
  __syncthreads();

  // Q A-tiles are loop invariant: hoist.
  v16bf qa0 = load_A(Qg + 0, HEAD, lane);
  v16bf qa1 = load_A(Qg + 32, HEAD, lane);

  // S tile row-block: wave w covers m-tiles [8w, 8w+8); B tiles from LDS.
  for (int mt = wave * 8; mt < wave * 8 + 8; ++mt) {
    v8f acc = {};
    acc = wmma_bf16(qa0, load_B_t(Ksh + mt * 16 * HEAD + 0, HEAD, lane), acc);
    acc = wmma_bf16(qa1, load_B_t(Ksh + mt * 16 * HEAD + 32, HEAD, lane), acc);
    int col = lane & 15, hi = lane >> 4;
#pragma unroll
    for (int r = 0; r < 8; ++r) {
      int nl = r + hi * 8;
      int m  = mt * 16 + col;
      float bias = affg[(size_t)(n0 + nl) * N_ROIS + m];  // log term from phase B
      Ssh[nl * 256 + m] = bias + acc[r] * 0.125f;         // 1/sqrt(64)
    }
  }
  __syncthreads();

  // softmax: 4 threads per row (quads are contiguous lanes -> xor 1,2 reduce)
  {
    int row = threadIdx.x >> 2, c0 = (threadIdx.x & 3) * 64;
    float* s = Ssh + row * 256 + c0;
    float mx = -3.4e38f;
    for (int c = 0; c < 64; ++c) mx = fmaxf(mx, s[c]);
    mx = fmaxf(mx, __shfl_xor(mx, 1));
    mx = fmaxf(mx, __shfl_xor(mx, 2));
    float sum = 0.f;
    for (int c = 0; c < 64; ++c) { float e = __expf(s[c] - mx); s[c] = e; sum += e; }
    sum += __shfl_xor(sum, 1);
    sum += __shfl_xor(sum, 2);
    float inv = 1.f / sum;
    float*  ao = affg + (size_t)(n0 + row) * N_ROIS + c0;
    __bf16* po = Psh + row * 256 + c0;
    for (int c = 0; c < 64; ++c) {
      float pval = s[c] * inv;
      ao[c] = pval;                // second reference output (aff_softmax)
      po[c] = (__bf16)pval;        // bf16 copy for the PV WMMA
    }
  }
  __syncthreads();

  // PV: out[16 x 128] = P[16 x 256] @ rf_f[256 x 128]; B from rfT (contiguous).
  const __bf16* VfT = rfT + (size_t)f * FEAT * N_ROIS;
  __bf16* ao = avb + ((size_t)(f * 16 + jj) * N_ROIS + n0) * FEAT;
  for (int dt = wave; dt < 8; dt += 2) {
    v8f acc = {};
#pragma unroll
    for (int kb = 0; kb < N_ROIS; kb += 32) {
      v16bf a = load_A(Psh + kb, 256, lane);
      v16bf b = load_B_t(VfT + (size_t)(dt * 16) * N_ROIS + kb, N_ROIS, lane);
      acc = wmma_bf16(a, b, acc);
    }
    int col = lane & 15, hi = lane >> 4;
#pragma unroll
    for (int r = 0; r < 8; ++r)
      ao[(size_t)(r + hi * 8) * FEAT + dt * 16 + col] = (__bf16)acc[r];
  }
}

// ---- phase D: 1x1 conv == per-f GEMM: Wc[128x2048] @ av_f[2048x256] ----
__global__ void __launch_bounds__(128) gemm_out(const __bf16* __restrict__ Wcb,
                                                const __bf16* __restrict__ avb,
                                                const float* __restrict__ bc,
                                                float* __restrict__ out0) {
  int wave = threadIdx.x >> 5, lane = threadIdx.x & 31;
  int id = blockIdx.x * 4 + wave;                      // 2048 tiles
  int nt = id & 15, ot = (id >> 4) & 7, f = id >> 7;
  int nb = nt * 16, ob = ot * 16;

  v8f acc = {};
  for (int kb = 0; kb < KC; kb += 32) {
    int j = kb >> 7, db = kb & 127;                    // k = j*128 + d
    v16bf a = load_A(Wcb + (size_t)ob * KC + kb, KC, lane);
    v16bf b = load_B_t(avb + ((size_t)(f * 16 + j) * N_ROIS + nb) * FEAT + db,
                       FEAT, lane);
    acc = wmma_bf16(a, b, acc);
  }
  int col = lane & 15, hi = lane >> 4;
  int n = nb + col;
#pragma unroll
  for (int r = 0; r < 8; ++r) {
    int o = ob + r + hi * 8;
    // linear_out[o][n][f] flat == output.reshape view
    out0[(size_t)o * (N_ROIS * NUM_FG) + (size_t)n * NUM_FG + f] = acc[r] + bc[o];
  }
}

extern "C" void kernel_launch(void* const* d_in, const int* in_sizes, int n_in,
                              void* d_out, int out_size, void* d_ws, size_t ws_size,
                              hipStream_t stream) {
  (void)in_sizes; (void)n_in; (void)out_size; (void)ws_size;
  const float* roi = (const float*)d_in[0];
  const float* pos = (const float*)d_in[1];
  const float* W1  = (const float*)d_in[2];
  const float* b1  = (const float*)d_in[3];
  const float* W2  = (const float*)d_in[4];
  const float* b2  = (const float*)d_in[5];
  const float* W3  = (const float*)d_in[6];
  const float* b3  = (const float*)d_in[7];
  const float* Wc  = (const float*)d_in[8];
  const float* bc  = (const float*)d_in[9];

  float* out0 = (float*)d_out;                                 // 524288 floats
  float* aff  = out0 + (size_t)DIM2 * N_ROIS * NUM_FG;         // 16777216 floats

  char* w = (char*)d_ws;
  __bf16* rfb  = (__bf16*)w; w += (size_t)NUM_FG * N_ROIS * FEAT * 2;
  __bf16* rfT  = (__bf16*)w; w += (size_t)NUM_FG * FEAT * N_ROIS * 2;
  __bf16* W2b  = (__bf16*)w; w += (size_t)DIM0 * FEAT * 2;
  __bf16* W3b  = (__bf16*)w; w += (size_t)DIM0 * FEAT * 2;
  __bf16* Wcb  = (__bf16*)w; w += (size_t)DIM2 * KC * 2;
  __bf16* qbuf = (__bf16*)w; w += (size_t)GROUPS * N_ROIS * HEAD * 2;
  __bf16* kbuf = (__bf16*)w; w += (size_t)GROUPS * N_ROIS * HEAD * 2;
  __bf16* avb  = (__bf16*)w; w += (size_t)GROUPS * N_ROIS * FEAT * 2;

  pack_rf <<<(NUM_FG * N_ROIS * FEAT) / 256, 256, 0, stream>>>(roi, rfb);
  pack_rfT<<<(NUM_FG * N_ROIS * FEAT) / 256, 256, 0, stream>>>(roi, rfT);
  cvt_bf16<<<(DIM0 * FEAT) / 256, 256, 0, stream>>>(W2, W2b, DIM0 * FEAT);
  cvt_bf16<<<(DIM0 * FEAT) / 256, 256, 0, stream>>>(W3, W3b, DIM0 * FEAT);
  cvt_bf16<<<(DIM2 * KC) / 256, 256, 0, stream>>>(Wc, Wcb, DIM2 * KC);
  bias_log<<<(NUM_FG * N_ROIS * N_ROIS) / 256, 256, 0, stream>>>(pos, W1, b1, aff);
  gemm_qk<<<16384 / 4, 128, 0, stream>>>(rfb, W2b, W3b, b2, b3, qbuf, kbuf);
  attn<<<GROUPS * 16, 64, 0, stream>>>(qbuf, kbuf, rfT, aff, avb);
  gemm_out<<<2048 / 4, 128, 0, stream>>>(Wcb, avb, bc, out0);
}
